// CSPLayer_48936857370783
// MI455X (gfx1250) — matrix operand order repaired and checked by reading, not compiled, not run on hardware
//
#include <hip/hip_runtime.h>

// CSPLayer fused GNN layer for MI455X (gfx1250, wave32, WMMA).
// N=50000 nodes, E=800000 edges, G=2048 graphs, H=128, EIN=265 (pad->288).
// Compute: ~85 GFLOP -> v_wmma_f32_16x16x32_bf16 (fp32 accum).
// Memory floor ~70MB (node_features L2-resident) -> matrix-core bound.

#define NN 50000
#define NE 800000
#define NG 2048
#define H  128

typedef __attribute__((ext_vector_type(16))) __bf16 v16bf;
typedef __attribute__((ext_vector_type(8)))  float  v8f;

union Frag {
    v16bf v;
    __bf16 h[16];
    uint4 q[2];
};

__device__ __forceinline__ float silu_f(float x) {
    return x / (1.0f + __expf(-x));
}

__device__ __forceinline__ void cvt8(float4 a, float4 b, __bf16* o) {
    o[0]=(__bf16)a.x; o[1]=(__bf16)a.y; o[2]=(__bf16)a.z; o[3]=(__bf16)a.w;
    o[4]=(__bf16)b.x; o[5]=(__bf16)b.y; o[6]=(__bf16)b.z; o[7]=(__bf16)b.w;
}

__device__ __forceinline__ void cvt8s(float4 a, float4 b, float s, __bf16* o) {
    o[0]=(__bf16)(a.x*s); o[1]=(__bf16)(a.y*s); o[2]=(__bf16)(a.z*s); o[3]=(__bf16)(a.w*s);
    o[4]=(__bf16)(b.x*s); o[5]=(__bf16)(b.y*s); o[6]=(__bf16)(b.z*s); o[7]=(__bf16)(b.w*s);
}

// ---- workspace layout (bytes) ----
#define WS_EW1 0          // 8 ntiles * 9 ktiles * 512 = 36864 bf16 -> 73728 B
#define WS_EW2 73728      // 8*4*512 = 16384 bf16 -> 32768 B
#define WS_NW1 106496     // 8*8*512 = 32768 bf16 -> 65536 B
#define WS_NW2 172032     // 16384 bf16 -> 32768 B
#define WS_CNT 204800     // NN floats -> 200000 B
#define WS_AGG 405504     // NN*H floats -> 25.6 MB

// Convert weight W[Kdim][128] (row-major) into WMMA B-fragment order:
// dst[(((n*Ktiles + t)*32 + lane)*16 + j)] = bf16(W[t*32 + (lane>=16?8:0) + (j<8?j:j+8)][n*16 + lane%16])
__device__ __forceinline__ void conv_frag(const float* __restrict__ W, int Kdim, int Ktiles,
                                          __bf16* __restrict__ dst, int i) {
    int j    = i & 15;
    int lane = (i >> 4) & 31;
    int tn   = i >> 9;
    int t    = tn % Ktiles;
    int n    = tn / Ktiles;
    int k    = t * 32 + ((lane >= 16) ? 8 : 0) + ((j < 8) ? j : j + 8);
    int col  = n * 16 + (lane & 15);
    float v  = (k < Kdim) ? W[k * 128 + col] : 0.0f;
    dst[i]   = (__bf16)v;
}

__global__ void prep_kernel(const float* __restrict__ ew1, const float* __restrict__ ew2,
                            const float* __restrict__ nw1, const float* __restrict__ nw2,
                            __bf16* __restrict__ fe1, __bf16* __restrict__ fe2,
                            __bf16* __restrict__ fn1, __bf16* __restrict__ fn2,
                            float* __restrict__ cnt, float* __restrict__ agg) {
    int tid = blockIdx.x * blockDim.x + threadIdx.x;
    int stride = gridDim.x * blockDim.x;
    for (int i = tid; i < NN * H; i += stride) agg[i] = 0.0f;
    for (int i = tid; i < NN; i += stride) cnt[i] = 0.0f;
    for (int i = tid; i < 8*9*512; i += stride) conv_frag(ew1, 265, 9, fe1, i);
    for (int i = tid; i < 8*4*512; i += stride) conv_frag(ew2, 128, 4, fe2, i);
    for (int i = tid; i < 8*8*512; i += stride) conv_frag(nw1, 256, 8, fn1, i);
    for (int i = tid; i < 8*4*512; i += stride) conv_frag(nw2, 128, 4, fn2, i);
}

// ================= edge pass =================
// one wave handles a 16-edge x 128-out tile; 9 K-tiles (hi|hj|lat+fd pad) then 4 K-tiles.
__global__ __launch_bounds__(256) void edge_kernel(
        const float* __restrict__ nf, const float* __restrict__ lat,
        const float* __restrict__ fd, const int* __restrict__ eidx,
        const int* __restrict__ e2g,
        const float* __restrict__ b1, const float* __restrict__ b2,
        const __bf16* __restrict__ wfrag,   // ew1 frags then ew2 frags (contiguous)
        float* __restrict__ agg, float* __restrict__ cnt) {
    extern __shared__ char smem[];
    __bf16* w1f = (__bf16*)smem;            // 36864
    __bf16* w2f = w1f + 36864;              // 16384
    __bf16* ebase = w2f + 16384;            // per-wave scratch

    // cooperative weight stage: 106496 B = 6656 uint4
    {
        const uint4* src = (const uint4*)wfrag;
        uint4* dst = (uint4*)smem;
        for (int i = threadIdx.x; i < 6656; i += blockDim.x) dst[i] = src[i];
    }
    __syncthreads();

    const int lane = threadIdx.x & 31;
    const int wave = threadIdx.x >> 5;
    const int m    = lane & 15;
    const int hi16 = lane >> 4;
    const int sub  = hi16 ? 8 : 0;
    __bf16* lds_e = ebase + wave * (16 * 128 + 32);
    int* lds_idx = (int*)(lds_e + 16 * 128);

    const int nwaves = (gridDim.x * blockDim.x) >> 5;
    int gwave = (blockIdx.x * blockDim.x + threadIdx.x) >> 5;
    const int ntiles = NE / 16;

    for (int tile = gwave; tile < ntiles; tile += nwaves) {
        int e  = tile * 16 + m;
        int i0 = eidx[e];
        int i1 = eidx[NE + e];
        int g  = e2g[e];
        const float* hrow = nf + (long)i0 * H;
        const float* jrow = nf + (long)i1 * H;
        if (hi16 == 0) lds_idx[m] = i0;

        float ex[9];
        #pragma unroll
        for (int q = 0; q < 6; ++q) ex[q] = lat[g * 6 + q];
        #pragma unroll
        for (int q = 0; q < 3; ++q) ex[6 + q] = fd[e * 3 + q];

        v8f acc[8];
        #pragma unroll
        for (int n = 0; n < 8; ++n)
            #pragma unroll
            for (int r = 0; r < 8; ++r) acc[n][r] = 0.0f;

        // GEMM1: [16x288] x [288x128]
        #pragma unroll
        for (int t = 0; t < 9; ++t) {
            Frag a;
            if (t < 8) {
                const float* row = (t < 4) ? hrow : jrow;
                int kb = (t & 3) * 32 + sub;
                float4 c0a = *(const float4*)(row + kb);
                float4 c0b = *(const float4*)(row + kb + 4);
                float4 c1a = *(const float4*)(row + kb + 16);
                float4 c1b = *(const float4*)(row + kb + 20);
                cvt8(c0a, c0b, a.h);
                cvt8(c1a, c1b, a.h + 8);
            } else {
                #pragma unroll
                for (int q = 0; q < 16; ++q) a.h[q] = (__bf16)0.0f;
                if (hi16 == 0) {
                    #pragma unroll
                    for (int q = 0; q < 8; ++q) a.h[q] = (__bf16)ex[q];   // K 256..263
                } else {
                    a.h[0] = (__bf16)ex[8];                               // K 264
                }
            }
            #pragma unroll
            for (int n = 0; n < 8; ++n) {
                Frag b;
                b.v = *(const v16bf*)(w1f + (((n * 9 + t) * 32 + lane) << 4));
                acc[n] = __builtin_amdgcn_wmma_f32_16x16x32_bf16(
                    false, a.v, false, b.v, (short)0, acc[n], false, false);
            }
        }

        // bias + SiLU, stash bf16 tile for transpose (C is col-per-lane, A needs row-per-lane)
        #pragma unroll
        for (int n = 0; n < 8; ++n) {
            float bb = b1[n * 16 + m];
            #pragma unroll
            for (int r = 0; r < 8; ++r) {
                float x = silu_f(acc[n][r] + bb);
                lds_e[(r + sub) * 128 + n * 16 + m] = (__bf16)x;
            }
        }

        v8f acc2[8];
        #pragma unroll
        for (int n = 0; n < 8; ++n)
            #pragma unroll
            for (int r = 0; r < 8; ++r) acc2[n][r] = 0.0f;

        // GEMM2: [16x128] x [128x128]
        #pragma unroll
        for (int t = 0; t < 4; ++t) {
            Frag a;
            int off = m * 128 + t * 32 + sub;
            a.q[0] = *(const uint4*)(lds_e + off);
            a.q[1] = *(const uint4*)(lds_e + off + 16);
            #pragma unroll
            for (int n = 0; n < 8; ++n) {
                Frag b;
                b.v = *(const v16bf*)(w2f + (((n * 4 + t) * 32 + lane) << 4));
                acc2[n] = __builtin_amdgcn_wmma_f32_16x16x32_bf16(
                    false, a.v, false, b.v, (short)0, acc2[n], false, false);
            }
        }

        // bias + SiLU + scatter-add (L2-resident fp32 atomics)
        #pragma unroll
        for (int n = 0; n < 8; ++n) {
            float bb = b2[n * 16 + m];
            int col = n * 16 + m;
            #pragma unroll
            for (int r = 0; r < 8; ++r) {
                float x = silu_f(acc2[n][r] + bb);
                int dst = lds_idx[r + sub];
                unsafeAtomicAdd(&agg[(long)dst * H + col], x);
            }
        }
        if (hi16 == 0) unsafeAtomicAdd(&cnt[i0], 1.0f);
    }
}

// ================= node pass =================
__global__ __launch_bounds__(256) void node_kernel(
        const float* __restrict__ nf,
        const float* __restrict__ b1, const float* __restrict__ b2,
        const __bf16* __restrict__ wfrag,   // nw1 frags then nw2 frags
        const float* __restrict__ agg, const float* __restrict__ cnt,
        float* __restrict__ out) {
    extern __shared__ char smem[];
    __bf16* w1f = (__bf16*)smem;            // 32768
    __bf16* w2f = w1f + 32768;              // 16384
    __bf16* ebase = w2f + 16384;

    {   // 98304 B = 6144 uint4
        const uint4* src = (const uint4*)wfrag;
        uint4* dst = (uint4*)smem;
        for (int i = threadIdx.x; i < 6144; i += blockDim.x) dst[i] = src[i];
    }
    __syncthreads();

    const int lane = threadIdx.x & 31;
    const int wave = threadIdx.x >> 5;
    const int m    = lane & 15;
    const int hi16 = lane >> 4;
    const int sub  = hi16 ? 8 : 0;
    __bf16* lds_e = ebase + wave * (16 * 128);

    const int nwaves = (gridDim.x * blockDim.x) >> 5;
    int gwave = (blockIdx.x * blockDim.x + threadIdx.x) >> 5;
    const int ntiles = NN / 16;

    for (int tile = gwave; tile < ntiles; tile += nwaves) {
        int node = tile * 16 + m;
        const float* row  = nf  + (long)node * H;
        const float* arow = agg + (long)node * H;
        float rinv = 1.0f / fmaxf(cnt[node], 1.0f);

        v8f acc[8];
        #pragma unroll
        for (int n = 0; n < 8; ++n)
            #pragma unroll
            for (int r = 0; r < 8; ++r) acc[n][r] = 0.0f;

        // GEMM1: [16x256] x [256x128]   (K tiles 0-3: node_features, 4-7: agg/cnt)
        #pragma unroll
        for (int t = 0; t < 8; ++t) {
            Frag a;
            int kb = (t & 3) * 32 + sub;
            if (t < 4) {
                float4 c0a = *(const float4*)(row + kb);
                float4 c0b = *(const float4*)(row + kb + 4);
                float4 c1a = *(const float4*)(row + kb + 16);
                float4 c1b = *(const float4*)(row + kb + 20);
                cvt8(c0a, c0b, a.h);
                cvt8(c1a, c1b, a.h + 8);
            } else {
                float4 c0a = *(const float4*)(arow + kb);
                float4 c0b = *(const float4*)(arow + kb + 4);
                float4 c1a = *(const float4*)(arow + kb + 16);
                float4 c1b = *(const float4*)(arow + kb + 20);
                cvt8s(c0a, c0b, rinv, a.h);
                cvt8s(c1a, c1b, rinv, a.h + 8);
            }
            #pragma unroll
            for (int n = 0; n < 8; ++n) {
                Frag b;
                b.v = *(const v16bf*)(w1f + (((n * 8 + t) * 32 + lane) << 4));
                acc[n] = __builtin_amdgcn_wmma_f32_16x16x32_bf16(
                    false, a.v, false, b.v, (short)0, acc[n], false, false);
            }
        }

        #pragma unroll
        for (int n = 0; n < 8; ++n) {
            float bb = b1[n * 16 + m];
            #pragma unroll
            for (int r = 0; r < 8; ++r) {
                float x = silu_f(acc[n][r] + bb);
                lds_e[(r + sub) * 128 + n * 16 + m] = (__bf16)x;
            }
        }

        v8f acc2[8];
        #pragma unroll
        for (int n = 0; n < 8; ++n)
            #pragma unroll
            for (int r = 0; r < 8; ++r) acc2[n][r] = 0.0f;

        #pragma unroll
        for (int t = 0; t < 4; ++t) {
            Frag a;
            int off = m * 128 + t * 32 + sub;
            a.q[0] = *(const uint4*)(lds_e + off);
            a.q[1] = *(const uint4*)(lds_e + off + 16);
            #pragma unroll
            for (int n = 0; n < 8; ++n) {
                Frag b;
                b.v = *(const v16bf*)(w2f + (((n * 4 + t) * 32 + lane) << 4));
                acc2[n] = __builtin_amdgcn_wmma_f32_16x16x32_bf16(
                    false, a.v, false, b.v, (short)0, acc2[n], false, false);
            }
        }

        // bias + SiLU + residual store
        #pragma unroll
        for (int n = 0; n < 8; ++n) {
            float bb = b2[n * 16 + m];
            int col = n * 16 + m;
            #pragma unroll
            for (int r = 0; r < 8; ++r) {
                float x = silu_f(acc2[n][r] + bb);
                long ridx = (long)(tile * 16 + r + sub) * H + col;
                out[ridx] = nf[ridx] + x;
            }
        }
    }
}

extern "C" void kernel_launch(void* const* d_in, const int* in_sizes, int n_in,
                              void* d_out, int out_size, void* d_ws, size_t ws_size,
                              hipStream_t stream) {
    const float* nf   = (const float*)d_in[0];
    // d_in[1] frac_coords: unused by reference
    const float* lat  = (const float*)d_in[2];
    const float* fd   = (const float*)d_in[3];
    const int*   eidx = (const int*)d_in[4];
    const int*   e2g  = (const int*)d_in[5];
    const float* ew1  = (const float*)d_in[6];
    const float* eb1  = (const float*)d_in[7];
    const float* ew2  = (const float*)d_in[8];
    const float* eb2  = (const float*)d_in[9];
    const float* nw1  = (const float*)d_in[10];
    const float* nb1  = (const float*)d_in[11];
    const float* nw2  = (const float*)d_in[12];
    const float* nb2  = (const float*)d_in[13];
    float* out = (float*)d_out;

    char* ws = (char*)d_ws;
    __bf16* fe1 = (__bf16*)(ws + WS_EW1);
    __bf16* fe2 = (__bf16*)(ws + WS_EW2);
    __bf16* fn1 = (__bf16*)(ws + WS_NW1);
    __bf16* fn2 = (__bf16*)(ws + WS_NW2);
    float* cnt = (float*)(ws + WS_CNT);
    float* agg = (float*)(ws + WS_AGG);

    prep_kernel<<<768, 256, 0, stream>>>(ew1, ew2, nw1, nw2, fe1, fe2, fn1, fn2, cnt, agg);

    // edge: LDS = 106496 (weights) + 8 waves * 4160 B (transpose scratch + idx) = 139776 B
    edge_kernel<<<1024, 256, 139776, stream>>>(nf, lat, fd, eidx, e2g, eb1, eb2,
                                               fe1 /* fe1+fe2 contiguous */, agg, cnt);

    // node: LDS = 98304 + 8 * 4096 = 131072 B
    node_kernel<<<256, 256, 131072, stream>>>(nf, nb1, nb2,
                                              fn1 /* fn1+fn2 contiguous */, agg, cnt, out);
}
